// MPNNBlock_38302518346486
// MI455X (gfx1250) — compile-verified, matrix operand order checked by reading, not run on hardware
//
#include <hip/hip_runtime.h>

#define H   128
#define HE  256
#define FF  512
#define NN  100000
#define NE  800000
#define INV_SCALE (1.0f/30.0f)
#define LN_EPS 1e-5f

typedef __attribute__((ext_vector_type(16))) __bf16 v16bf;
typedef __attribute__((ext_vector_type(8)))  float  v8f;

union FragU { v16bf v; unsigned u[8]; };

#define WMMA_BF16(a,b,c) \
  __builtin_amdgcn_wmma_f32_16x16x32_bf16(false,(a),false,(b),(short)0,(c),false,false)

// float -> bf16 (round to nearest even)
__device__ __forceinline__ unsigned short f2bf(float x) {
  unsigned u = __float_as_uint(x);
  u += 0x7FFFu + ((u >> 16) & 1u);
  return (unsigned short)(u >> 16);
}

// K offset inside a 16-bit A/B fragment: VGPR v (0..7), lane-half hf (0/1)
__device__ __forceinline__ int kfrag(int v, int hf) {
  return ((v >> 2) << 4) + (hf << 3) + ((v & 3) << 1);
}

// Gather a 16x32 bf16 fragment (A from row-major tile, or B from W^T row).
__device__ __forceinline__ v16bf load_frag(const unsigned short* p, int hf) {
  FragU f;
#pragma unroll
  for (int v = 0; v < 8; ++v)
    f.u[v] = *(const unsigned*)(p + kfrag(v, hf));
  return f.v;
}

// ---------------- kernel 0: zero dh accumulator (d_out) ----------------
__global__ void zero_f4(float4* __restrict__ p, int n4) {
  int i = blockIdx.x * 256 + threadIdx.x;
  if (i < n4) p[i] = make_float4(0.f, 0.f, 0.f, 0.f);
}

// ------------- kernel 1: weights -> bf16, transposed [N][K] ------------
// ws layout (u16 elems): W1t @0 (128x256), W2t @32768 (128x128),
// W3t @49152 (128x128), D1t @65536 (512x128), D2t @131072 (128x512)
__global__ void wconv(const float* __restrict__ W1, const float* __restrict__ W2,
                      const float* __restrict__ W3, const float* __restrict__ D1,
                      const float* __restrict__ D2, unsigned short* __restrict__ ws) {
  int idx = blockIdx.x * 256 + threadIdx.x;
  if (idx >= 196608) return;
  float val;
  if (idx < 32768)        { int o = idx;          int n = o / HE, k = o % HE; val = W1[(size_t)k * H  + n]; }
  else if (idx < 49152)   { int o = idx - 32768;  int n = o / H,  k = o % H;  val = W2[(size_t)k * H  + n]; }
  else if (idx < 65536)   { int o = idx - 49152;  int n = o / H,  k = o % H;  val = W3[(size_t)k * H  + n]; }
  else if (idx < 131072)  { int o = idx - 65536;  int n = o / H,  k = o % H;  val = D1[(size_t)k * FF + n]; }
  else                    { int o = idx - 131072; int n = o / FF, k = o % FF; val = D2[(size_t)k * H  + n]; }
  ws[idx] = f2bf(val);
}

// ---------- kernel 2: edge MLP (3 layers, WMMA) + atomic scatter -------
// 64 edges per block; 8 waves = 4 row-tiles x 2 column-halves (64 cols each)
__global__ void __launch_bounds__(256)
edge_mlp_scatter(const float* __restrict__ hE, const int* __restrict__ srcIdx,
                 const unsigned short* __restrict__ W1t,
                 const unsigned short* __restrict__ W2t,
                 const unsigned short* __restrict__ W3t,
                 const float* __restrict__ b1, const float* __restrict__ b2,
                 const float* __restrict__ b3, float* __restrict__ dh) {
  __shared__ __align__(16) unsigned short eA[64 * HE];   // 32 KB
  __shared__ __align__(16) unsigned short m1[64 * H];    // 16 KB
  __shared__ __align__(16) unsigned short m2[64 * H];    // 16 KB

  const int t     = threadIdx.x;
  const int ebase = blockIdx.x * 64;

  // stage 64x256 edge tile as bf16 (float4 loads -> packed uint2 LDS stores)
  {
    const float4* s4 = (const float4*)(hE + (size_t)ebase * HE);
    uint2* d2 = (uint2*)eA;
#pragma unroll
    for (int i = 0; i < 16; ++i) {
      int q = t + i * 256;               // 4096 quads total
      float4 v = s4[q];
      uint2 p;
      p.x = (unsigned)f2bf(v.x) | ((unsigned)f2bf(v.y) << 16);
      p.y = (unsigned)f2bf(v.z) | ((unsigned)f2bf(v.w) << 16);
      d2[q] = p;
    }
  }
  __syncthreads();

  const int wave = t >> 5;
  const int lane = t & 31;
  const int hf   = lane >> 4;     // lane half
  const int nloc = lane & 15;     // N (and A-row M) within tile
  const int rt   = wave >> 1;     // row-tile 0..3 (16 edges each)
  const int nb   = (wave & 1) * 64;  // this wave's 64 output cols
  const int arow = rt * 16 + nloc;   // A-fragment row in LDS tiles

  v8f acc[4];

  // ---- layer 1: [64x256] @ W1  (K = 256)
#pragma unroll
  for (int i = 0; i < 4; ++i) acc[i] = (v8f){};
#pragma unroll
  for (int ks = 0; ks < HE / 32; ++ks) {
    v16bf a = load_frag(eA + arow * HE + ks * 32, hf);
#pragma unroll
    for (int i = 0; i < 4; ++i) {
      v16bf b = load_frag(W1t + (size_t)(nb + i * 16 + nloc) * HE + ks * 32, hf);
      acc[i] = WMMA_BF16(a, b, acc[i]);
    }
  }
#pragma unroll
  for (int i = 0; i < 4; ++i) {
    int n = nb + i * 16 + nloc;
    float bb = b1[n];
#pragma unroll
    for (int j = 0; j < 8; ++j) {
      int M = rt * 16 + j + 8 * hf;
      float x = acc[i][j] + bb; x = x > 0.f ? x : 0.f;
      m1[M * H + n] = f2bf(x);
    }
  }
  __syncthreads();

  // ---- layer 2: [64x128] @ W2  (K = 128)
#pragma unroll
  for (int i = 0; i < 4; ++i) acc[i] = (v8f){};
#pragma unroll
  for (int ks = 0; ks < H / 32; ++ks) {
    v16bf a = load_frag(m1 + arow * H + ks * 32, hf);
#pragma unroll
    for (int i = 0; i < 4; ++i) {
      v16bf b = load_frag(W2t + (size_t)(nb + i * 16 + nloc) * H + ks * 32, hf);
      acc[i] = WMMA_BF16(a, b, acc[i]);
    }
  }
#pragma unroll
  for (int i = 0; i < 4; ++i) {
    int n = nb + i * 16 + nloc;
    float bb = b2[n];
#pragma unroll
    for (int j = 0; j < 8; ++j) {
      int M = rt * 16 + j + 8 * hf;
      float x = acc[i][j] + bb; x = x > 0.f ? x : 0.f;
      m2[M * H + n] = f2bf(x);
    }
  }
  __syncthreads();

  // ---- layer 3 (no relu) + scatter-add into dh[src[e]]
#pragma unroll
  for (int i = 0; i < 4; ++i) acc[i] = (v8f){};
#pragma unroll
  for (int ks = 0; ks < H / 32; ++ks) {
    v16bf a = load_frag(m2 + arow * H + ks * 32, hf);
#pragma unroll
    for (int i = 0; i < 4; ++i) {
      v16bf b = load_frag(W3t + (size_t)(nb + i * 16 + nloc) * H + ks * 32, hf);
      acc[i] = WMMA_BF16(a, b, acc[i]);
    }
  }
#pragma unroll
  for (int j = 0; j < 8; ++j) {
    int e   = ebase + rt * 16 + j + 8 * hf;
    int row = srcIdx[e];                     // edge_idx[0][e]
    float* drow = dh + (size_t)row * H;
#pragma unroll
    for (int i = 0; i < 4; ++i) {
      int n = nb + i * 16 + nloc;
      atomicAdd(drow + n, acc[i][j] + b3[n]);
    }
  }
}

// ------ kernel 3: LN1 -> FFN (WMMA) -> residual -> LN2 -> d_out --------
// 32 rows per block; 8 waves = 2 row-tiles x 4 column-groups
__global__ void __launch_bounds__(256)
node_update(const float* __restrict__ hV,
            const float* __restrict__ g1, const float* __restrict__ be1,
            const float* __restrict__ g2, const float* __restrict__ be2,
            const unsigned short* __restrict__ D1t,
            const unsigned short* __restrict__ D2t,
            const float* __restrict__ db1, const float* __restrict__ db2,
            float* __restrict__ out /* in: dh ; out: result */) {
  __shared__ __align__(16) unsigned short hbf[32 * H];   // 8 KB
  __shared__ __align__(16) unsigned short u[32 * FF];    // 32 KB
  __shared__ float hres[32 * H];                         // 16 KB
  __shared__ float yb[32 * H];                           // 16 KB
  __shared__ float ps[32 * 8], ps2[32 * 8];
  __shared__ float mn[32], rs[32];

  const int t     = threadIdx.x;
  const int rbase = blockIdx.x * 32;
  const int r     = t >> 3;        // 0..31: row in tile
  const int seg   = t & 7;         // 8 threads per row
  const int cseg  = seg * 16;
  const size_t go = (size_t)(rbase + r) * H;

  // x = hV + dh/SCALE ; stats pass
  float s = 0.f, s2 = 0.f;
#pragma unroll
  for (int i = 0; i < 16; ++i) {
    float x = hV[go + cseg + i] + out[go + cseg + i] * INV_SCALE;
    hres[r * H + cseg + i] = x;
    s += x; s2 += x * x;
  }
  ps[r * 8 + seg] = s; ps2[r * 8 + seg] = s2;
  __syncthreads();
  if (seg == 0) {
    float a = 0.f, b = 0.f;
#pragma unroll
    for (int k = 0; k < 8; ++k) { a += ps[r * 8 + k]; b += ps2[r * 8 + k]; }
    float mean = a * (1.f / H);
    mn[r] = mean;
    rs[r] = rsqrtf(b * (1.f / H) - mean * mean + LN_EPS);
  }
  __syncthreads();
  {
    float mean = mn[r], rstd = rs[r];
#pragma unroll
    for (int i = 0; i < 16; ++i) {
      int cc = cseg + i;
      float hn = (hres[r * H + cc] - mean) * rstd * g1[cc] + be1[cc];
      hres[r * H + cc] = hn;
      hbf[r * H + cc]  = f2bf(hn);
    }
  }
  __syncthreads();

  const int wave = t >> 5;
  const int lane = t & 31;
  const int hf   = lane >> 4;
  const int nloc = lane & 15;
  const int rt   = wave >> 2;            // row-tile 0..1
  const int arow = rt * 16 + nloc;

  // ---- FFN layer 1: [32x128] @ D1 -> this wave's 128 of 512 cols
  const int nb1 = (wave & 3) * 128;
  v8f acc[8];
#pragma unroll
  for (int i = 0; i < 8; ++i) acc[i] = (v8f){};
#pragma unroll
  for (int ks = 0; ks < H / 32; ++ks) {
    v16bf a = load_frag(hbf + arow * H + ks * 32, hf);
#pragma unroll
    for (int i = 0; i < 8; ++i) {
      v16bf b = load_frag(D1t + (size_t)(nb1 + i * 16 + nloc) * H + ks * 32, hf);
      acc[i] = WMMA_BF16(a, b, acc[i]);
    }
  }
#pragma unroll
  for (int i = 0; i < 8; ++i) {
    int n = nb1 + i * 16 + nloc;
    float bb = db1[n];
#pragma unroll
    for (int j = 0; j < 8; ++j) {
      int M = rt * 16 + j + 8 * hf;
      float x = acc[i][j] + bb; x = x > 0.f ? x : 0.f;
      u[M * FF + n] = f2bf(x);
    }
  }
  __syncthreads();

  // ---- FFN layer 2: [32x512] @ D2 -> this wave's 32 of 128 cols
  const int nb2 = (wave & 3) * 32;
  v8f d0 = {}, d1 = {};
#pragma unroll
  for (int ks = 0; ks < FF / 32; ++ks) {
    v16bf a  = load_frag(u + arow * FF + ks * 32, hf);
    v16bf bA = load_frag(D2t + (size_t)(nb2 + nloc)      * FF + ks * 32, hf);
    v16bf bB = load_frag(D2t + (size_t)(nb2 + 16 + nloc) * FF + ks * 32, hf);
    d0 = WMMA_BF16(a, bA, d0);
    d1 = WMMA_BF16(a, bB, d1);
  }
  {
    float bb0 = db2[nb2 + nloc], bb1 = db2[nb2 + 16 + nloc];
#pragma unroll
    for (int j = 0; j < 8; ++j) {
      int M = rt * 16 + j + 8 * hf;
      yb[M * H + nb2 + nloc]      = hres[M * H + nb2 + nloc]      + d0[j] + bb0;
      yb[M * H + nb2 + 16 + nloc] = hres[M * H + nb2 + 16 + nloc] + d1[j] + bb1;
    }
  }
  __syncthreads();

  // ---- LN2 + store
  s = 0.f; s2 = 0.f;
#pragma unroll
  for (int i = 0; i < 16; ++i) {
    float x = yb[r * H + cseg + i];
    s += x; s2 += x * x;
  }
  ps[r * 8 + seg] = s; ps2[r * 8 + seg] = s2;
  __syncthreads();
  if (seg == 0) {
    float a = 0.f, b = 0.f;
#pragma unroll
    for (int k = 0; k < 8; ++k) { a += ps[r * 8 + k]; b += ps2[r * 8 + k]; }
    float mean = a * (1.f / H);
    mn[r] = mean;
    rs[r] = rsqrtf(b * (1.f / H) - mean * mean + LN_EPS);
  }
  __syncthreads();
  {
    float mean = mn[r], rstd = rs[r];
#pragma unroll
    for (int i = 0; i < 16; ++i) {
      int cc = cseg + i;
      out[go + cc] = (yb[r * H + cc] - mean) * rstd * g2[cc] + be2[cc];
    }
  }
}

extern "C" void kernel_launch(void* const* d_in, const int* in_sizes, int n_in,
                              void* d_out, int out_size, void* d_ws, size_t ws_size,
                              hipStream_t stream) {
  const float* hV  = (const float*)d_in[0];
  const float* hE  = (const float*)d_in[1];
  const int*   eix = (const int*)d_in[2];   // [2, NE]; row 0 = src
  const float* W1  = (const float*)d_in[3];
  const float* b1  = (const float*)d_in[4];
  const float* W2  = (const float*)d_in[5];
  const float* b2  = (const float*)d_in[6];
  const float* W3  = (const float*)d_in[7];
  const float* b3  = (const float*)d_in[8];
  const float* D1  = (const float*)d_in[9];
  const float* db1 = (const float*)d_in[10];
  const float* D2  = (const float*)d_in[11];
  const float* db2 = (const float*)d_in[12];
  const float* g1  = (const float*)d_in[13];
  const float* be1 = (const float*)d_in[14];
  const float* g2  = (const float*)d_in[15];
  const float* be2 = (const float*)d_in[16];

  unsigned short* ws = (unsigned short*)d_ws;
  float* out = (float*)d_out;

  // 1) zero dh accumulator (reuse d_out)
  int n4 = (NN * H) / 4;
  zero_f4<<<(n4 + 255) / 256, 256, 0, stream>>>((float4*)out, n4);
  // 2) bf16 transposed weights into workspace
  wconv<<<(196608 + 255) / 256, 256, 0, stream>>>(W1, W2, W3, D1, D2, ws);
  // 3) edge MLP + scatter (src = eix[0..NE))
  edge_mlp_scatter<<<NE / 64, 256, 0, stream>>>(hE, eix, ws, ws + 32768, ws + 49152,
                                                b1, b2, b3, out);
  // 4) node update (LN -> FFN -> LN)
  node_update<<<NN / 32, 256, 0, stream>>>(hV, g1, be1, g2, be2,
                                           ws + 65536, ws + 131072, db1, db2, out);
}
